// SNNonLocalBlock_87797721465209
// MI455X (gfx1250) — matrix operand order repaired and checked by reading, not compiled, not run on hardware
//
#include <hip/hip_runtime.h>

typedef __attribute__((ext_vector_type(16))) _Float16 v16h;
typedef __attribute__((ext_vector_type(8)))  float    v8f;

#define USE_ASYNC_LDS 1

union frag16 { v16h h; uint4 u4[2]; unsigned u32[8]; _Float16 e[16]; };

__device__ __forceinline__ uint4 ld_u4(const _Float16* p) {
    return *reinterpret_cast<const uint4*>(p);
}

__device__ __forceinline__ v8f wmma_f16(v16h a, v16h b, v8f c) {
    // D = A(16x32 f16) * B(32x16 f16) + C(16x16 f32)
    return __builtin_amdgcn_wmma_f32_16x16x32_f16(false, a, false, b, (short)0, c,
                                                  false, false);
}

__device__ __forceinline__ unsigned pk2(float a, float b) {
    union { _Float16 h[2]; unsigned u; } v;
    v.h[0] = (_Float16)a; v.h[1] = (_Float16)b;
    return v.u;
}

__device__ __forceinline__ uint4 pack8(v8f a) {
    union { uint4 u; _Float16 h[8]; } r;
#pragma unroll
    for (int i = 0; i < 8; i++) r.h[i] = (_Float16)a[i];
    return r.u;
}

__device__ __forceinline__ uint4 hmax8(uint4 a, uint4 b) {
    union U { uint4 u; _Float16 h[8]; } x, y, r;
    x.u = a; y.u = b;
#pragma unroll
    for (int i = 0; i < 8; i++) r.h[i] = (x.h[i] > y.h[i]) ? x.h[i] : y.h[i];
    return r.u;
}

// A-fragment load: lane<16 holds K{klo..klo+7, 16+klo..+7}, klo = hi*8
__device__ __forceinline__ void load_a_f32(frag16& A, const float* xp) {
    const float4* x4 = reinterpret_cast<const float4*>(xp);
    union { float4 f4[4]; float f[16]; } xf;
    xf.f4[0] = x4[0]; xf.f4[1] = x4[1];   // K klo..klo+7
    xf.f4[2] = x4[4]; xf.f4[3] = x4[5];   // K 16+klo..
#pragma unroll
    for (int i = 0; i < 16; i++) A.e[i] = (_Float16)xf.f[i];
}

// ---------------------------------------------------------------------------
// Problem constants: B=16, H=W=64, C=256, L=4096, Ld=1024, d=32, dv=128
// Combined projection weight Wc = [w_theta | w_phi | w_g] : [256, 192]
//
// Workspace layout (bytes)
#define WC_OFF    0u          // 12*8*32*16 f16 = 98304 B (B-fragment swizzled)
#define WO_OFF    98304u      // 16*4*32*16 f16 = 65536 B (B-fragment swizzled)
#define THETA_OFF 163840u     // [B*L][32]  f16 = 4 MB
#define PHIF_OFF  4358144u    // [B*L][32]  f16 = 4 MB (full-res phi)
#define PHIP_OFF  8552448u    // [B*Ld][32] f16 = 1 MB (pooled phi)
#define GT_OFF    9601024u    // [B*128][4096] f16 = 16 MB (transposed full-res g)
#define GTP_OFF   26378240u   // [B*128][1024] f16 = 4 MB (transposed pooled g)

// ---------------------------------------------------------------------------
// Kernel 0: swizzle f32 weights -> f16 B-matrix fragment layout.
// B layout: lane n<16 holds K=0..7,16..23 (elem order), lane n+16 holds K=8..15,24..31.
// ---------------------------------------------------------------------------
__global__ __launch_bounds__(256) void swizzle_w_kernel(
    const float* __restrict__ wt, const float* __restrict__ wph,
    const float* __restrict__ wg, const float* __restrict__ wo,
    _Float16* __restrict__ Wc, _Float16* __restrict__ Wo) {
    int tid = blockIdx.x * 256 + threadIdx.x;
    if (tid < 49152) {                       // combined [256,192] weight
        int e    = tid & 15;
        int lane = (tid >> 4) & 31;
        int kt   = (tid >> 9) & 7;
        int nt   = tid >> 12;                // 0..11
        int n = nt * 16 + (lane & 15);       // output col 0..191
        int k = kt * 32 + ((lane >> 4) << 3) + (e & 7) + ((e >> 3) << 4);   // 0..255
        float v;
        if (n < 32)       v = wt[k * 32 + n];
        else if (n < 64)  v = wph[k * 32 + (n - 32)];
        else              v = wg[k * 128 + (n - 64)];
        Wc[tid] = (_Float16)v;
    } else if (tid < 81920) {                // w_o [128,256]
        int t    = tid - 49152;
        int e    = t & 15;
        int lane = (t >> 4) & 31;
        int kt   = (t >> 9) & 3;
        int nt   = t >> 11;                  // 0..15
        int n = nt * 16 + (lane & 15);       // 0..255
        int k = kt * 32 + ((lane >> 4) << 3) + (e & 7) + ((e >> 3) << 4);   // 0..127
        Wo[t] = (_Float16)wo[k * 256 + n];
    }
}

// ---------------------------------------------------------------------------
// Kernel 1: fused projection GEMM.  Per wave: 2 M-tiles x 6 N-tiles so every
// B-fragment feeds two WMMAs, with an explicit next-B double buffer to break
// the load->wmma serialization seen in round-0 asm.  8 waves = 4 M-pairs x
// 2 N-halves -> 128 rows/WG, 512 WGs, 96 WMMA/wave.
// ---------------------------------------------------------------------------
__global__ __launch_bounds__(256) void proj_kernel(
    const float* __restrict__ x, const _Float16* __restrict__ Wc,
    _Float16* __restrict__ theta, _Float16* __restrict__ phif,
    _Float16* __restrict__ gt) {
    const int wave = threadIdx.x >> 5, lane = threadIdx.x & 31;
    const int m = lane & 15, hi = lane >> 4;
    const int mp = wave >> 1;                 // M-pair 0..3
    const int nh = wave & 1;                  // N-half 0..1 (N-tiles nh*6..+5)
    const int rowbase = blockIdx.x * 128 + mp * 32;  // global row (b*4096+loc)
    const int b = rowbase >> 12, loc = rowbase & 4095;

    v8f zero8 = {};
    v8f acc[2][6];
#pragma unroll
    for (int i = 0; i < 2; i++)
#pragma unroll
        for (int j = 0; j < 6; j++) acc[i][j] = zero8;

    const float* xrow0 = x + (size_t)(rowbase + m) * 256;
    const float* xrow1 = x + (size_t)(rowbase + 16 + m) * 256;

#pragma unroll
    for (int kt = 0; kt < 8; kt++) {
        frag16 A0, A1;
        load_a_f32(A0, xrow0 + kt * 32 + hi * 8);
        load_a_f32(A1, xrow1 + kt * 32 + hi * 8);
        // Wc entry index = ((nt*8+kt)*32+lane)*16 ; step per nt = 4096 elements
        const _Float16* wbase = Wc + ((size_t)((nh * 48 + kt) * 32 + lane)) * 16;
        frag16 Bc, Bn;
        Bc.u4[0] = ld_u4(wbase);
        Bc.u4[1] = ld_u4(wbase + 8);
#pragma unroll
        for (int j = 0; j < 6; j++) {
            if (j < 5) {   // prefetch next B while current pair computes
                const _Float16* wp = wbase + (size_t)(j + 1) * 4096;
                Bn.u4[0] = ld_u4(wp);
                Bn.u4[1] = ld_u4(wp + 8);
            }
            acc[0][j] = wmma_f16(A0.h, Bc.h, acc[0][j]);
            acc[1][j] = wmma_f16(A1.h, Bc.h, acc[1][j]);
            if (j < 5) Bc = Bn;
        }
    }

    // Stores. D layout: lane n: col = n%16, rows r + hi*8 (r = VGPR index).
#pragma unroll
    for (int j = 0; j < 6; j++) {
        const int nt = nh * 6 + j;
#pragma unroll
        for (int i = 0; i < 2; i++) {
            if (nt < 2) {            // theta, row-major [row][32]
#pragma unroll
                for (int r = 0; r < 8; r++)
                    theta[(size_t)(rowbase + i * 16 + hi * 8 + r) * 32 +
                          nt * 16 + m] = (_Float16)acc[i][j][r];
            } else if (nt < 4) {     // phi (full-res), row-major [row][32]
#pragma unroll
                for (int r = 0; r < 8; r++)
                    phif[(size_t)(rowbase + i * 16 + hi * 8 + r) * 32 +
                         (nt - 2) * 16 + m] = (_Float16)acc[i][j][r];
            } else {                 // g transposed [b][ch][loc], 16B per lane
                const int ch = (nt - 4) * 16 + m;
                _Float16* dst = gt + ((size_t)(b * 128 + ch)) * 4096 + loc +
                                i * 16 + hi * 8;
                *reinterpret_cast<uint4*>(dst) = pack8(acc[i][j]);
            }
        }
    }
}

// ---------------------------------------------------------------------------
// Kernel 2: 2x2 maxpool of phi, row-major in/out. 65536 threads.
// ---------------------------------------------------------------------------
__global__ __launch_bounds__(256) void pool_phi_kernel(
    const _Float16* __restrict__ phif, _Float16* __restrict__ phip) {
    int tid = blockIdx.x * 256 + threadIdx.x;   // 0..65535
    int kg = tid >> 2;                          // b*1024 + key
    int c8 = (tid & 3) * 8;
    int b = kg >> 10, k = kg & 1023;
    int hp = k >> 5, wp = k & 31;
    size_t l00 = (size_t)(b * 4096 + hp * 128 + wp * 2);
    uint4 a = ld_u4(phif + l00 * 32 + c8);
    uint4 bb = ld_u4(phif + (l00 + 1) * 32 + c8);
    uint4 c = ld_u4(phif + (l00 + 64) * 32 + c8);
    uint4 d = ld_u4(phif + (l00 + 65) * 32 + c8);
    uint4 r = hmax8(hmax8(a, bb), hmax8(c, d));
    *reinterpret_cast<uint4*>(phip + (size_t)kg * 32 + c8) = r;
}

// ---------------------------------------------------------------------------
// Kernel 3: 2x2 maxpool of transposed g along loc within each channel row.
// gt [b*128+ch][4096] -> gtp [b*128+ch][1024]. 262144 threads, coalesced.
// ---------------------------------------------------------------------------
__global__ __launch_bounds__(256) void pool_g_kernel(
    const _Float16* __restrict__ gt, _Float16* __restrict__ gtp) {
    int tid = blockIdx.x * 256 + threadIdx.x;   // 0..262143
    int r = tid >> 2;                           // (b*128+ch)*32 + hp
    int chg = r >> 5;                           // b*128 + ch
    int hp = r & 31;
    int wp8 = (tid & 3) * 8;                    // 8 pooled outputs / thread
    size_t base = (size_t)chg * 4096 + hp * 128 + wp8 * 2;
    uint4 a0 = ld_u4(gt + base),      a1 = ld_u4(gt + base + 8);
    uint4 b0 = ld_u4(gt + base + 64), b1 = ld_u4(gt + base + 72);
    uint4 m0 = hmax8(a0, b0), m1 = hmax8(a1, b1);   // vertical max
    union { uint4 u[2]; _Float16 h[16]; } mm; mm.u[0] = m0; mm.u[1] = m1;
    union { uint4 u; _Float16 h[8]; } o;
#pragma unroll
    for (int i = 0; i < 8; i++) {
        _Float16 x0 = mm.h[2 * i], x1 = mm.h[2 * i + 1];
        o.h[i] = (x0 > x1) ? x0 : x1;           // horizontal max
    }
    *reinterpret_cast<uint4*>(gtp + (size_t)chg * 1024 + hp * 32 + wp8) = o.u;
}

// ---------------------------------------------------------------------------
// Kernel 4: fused flash-attention + output projection + residual.
// Grid (32 qblocks, 16 batches), 256 threads. Wave w handles 16 queries.
//  S^T = phi_tile(16k x 32) x theta^T(32 x 16q)   [2 WMMA per 32-key block]
//  O^T += gT_tile(16c x 32k) x P^T(32k x 16q)     [8 WMMA per 32-key block]
//  epilogue: Y = attn_g(16q x 128) x w_o(128 x 256) [4x16 WMMA], +b_o, residual.
// g key-blocks staged to LDS via GLOBAL_LOAD_ASYNC_TO_LDS_B128 (ASYNCcnt).
// ---------------------------------------------------------------------------
__global__ __launch_bounds__(256) void attn_kernel(
    const float* __restrict__ x, const _Float16* __restrict__ theta,
    const _Float16* __restrict__ phip, const _Float16* __restrict__ gtp,
    const _Float16* __restrict__ Wo, const float* __restrict__ b_o,
    const float* __restrict__ sigma_p, float* __restrict__ out) {
    __shared__ _Float16 gT[128 * 40];        // [128 ch][32 keys] pad-to-40
    __shared__ _Float16 aO[8 * 16 * 136];    // per-wave [16 q][128 ch] pad-to-136

    const int b = blockIdx.y;
    const int qb = blockIdx.x;
    const int wave = threadIdx.x >> 5, lane = threadIdx.x & 31;
    const int m = lane & 15, hi = lane >> 4;
    const int qbase = qb * 128 + wave * 16;
    const v8f zero8 = {};
    const float sigma = sigma_p[0];

    // theta as B-matrix: lane n holds col q=n%16, K(ch) 0..7,16..23 / 8..15,24..31
    frag16 Bq;
    {
        const _Float16* tp =
            theta + ((size_t)(b * 4096 + qbase + m)) * 32 + hi * 8;
        Bq.u4[0] = ld_u4(tp);
        Bq.u4[1] = ld_u4(tp + 16);
    }

    // Per-thread async staging addresses (advance by 32 keys = 64 B per block)
    const int sch = threadIdx.x >> 1, shalf = threadIdx.x & 1;
    const _Float16* gsrc =
        gtp + ((size_t)(b * 128 + sch)) * 1024 + shalf * 16;
    _Float16* gdst = &gT[sch * 40 + shalf * 16];
#if USE_ASYNC_LDS
    const unsigned lds_off = (unsigned)(uintptr_t)gdst;   // wave-relative LDS addr
#endif

    v8f O[8];
#pragma unroll
    for (int i = 0; i < 8; i++) O[i] = zero8;
    float m_run = -3.0e38f, l_run = 0.0f;

    for (int kb = 0; kb < 32; kb++) {
        __syncthreads();   // previous block's gT reads complete

        // phi A-fragments for keys kb*32..+31 (L2-resident) -- issue early
        frag16 PA0, PA1;
        {
            const _Float16* pp =
                phip + ((size_t)(b * 1024 + kb * 32 + m)) * 32 + hi * 8;
            PA0.u4[0] = ld_u4(pp);        PA0.u4[1] = ld_u4(pp + 16);
            PA1.u4[0] = ld_u4(pp + 512);  PA1.u4[1] = ld_u4(pp + 528);
        }

        // cooperative stage of g^T key-block: [128 ch][32 keys] (8 KB)
        {
            const _Float16* src = gsrc + kb * 32;
#if USE_ASYNC_LDS
            unsigned long long ga = (unsigned long long)(uintptr_t)src;
            asm volatile(
                "global_load_async_to_lds_b128 %0, %1, off\n\t"
                "global_load_async_to_lds_b128 %0, %1, off offset:16\n\t"
                "s_wait_asynccnt 0x0"
                :: "v"(lds_off), "v"(ga) : "memory");
#else
            uint4 u0 = ld_u4(src), u1 = ld_u4(src + 8);
            *reinterpret_cast<uint4*>(gdst) = u0;
            *reinterpret_cast<uint4*>(gdst + 8) = u1;
#endif
        }
        __syncthreads();

        v8f S0 = wmma_f16(PA0.h, Bq.h, zero8);   // [16 keys, 16 q]
        v8f S1 = wmma_f16(PA1.h, Bq.h, zero8);

        // online softmax: lane q / q+16 jointly own 32 key scores of query q
        float tmax = S0[0];
#pragma unroll
        for (int r = 1; r < 8; r++) tmax = fmaxf(tmax, S0[r]);
#pragma unroll
        for (int r = 0; r < 8; r++) tmax = fmaxf(tmax, S1[r]);
        tmax = fmaxf(tmax, __shfl_xor(tmax, 16, 32));
        float mnew = fmaxf(m_run, tmax);
        float scale = __expf(m_run - mnew);
        float p0[8], p1[8], lsum = 0.0f;
#pragma unroll
        for (int r = 0; r < 8; r++) {
            p0[r] = __expf(S0[r] - mnew);
            p1[r] = __expf(S1[r] - mnew);
            lsum += p0[r] + p1[r];
        }
        lsum += __shfl_xor(lsum, 16, 32);
        l_run = l_run * scale + lsum;
        m_run = mnew;
#pragma unroll
        for (int ct = 0; ct < 8; ct++) O[ct] = O[ct] * scale;

        // Build P^T B-fragment (32 keys x 16 q): one lane^16 half-exchange
        unsigned pk0[4], pk1[4];
#pragma unroll
        for (int i = 0; i < 4; i++) {
            pk0[i] = pk2(p0[2 * i], p0[2 * i + 1]);   // own S0 keys
            pk1[i] = pk2(p1[2 * i], p1[2 * i + 1]);   // own S1 keys
        }
        frag16 Pf;
#pragma unroll
        for (int i = 0; i < 4; i++) {
            unsigned send = hi ? pk0[i] : pk1[i];
            unsigned rcv = __shfl_xor(send, 16, 32);
            Pf.u32[i]     = hi ? rcv    : pk0[i];   // K 0..7  / 16..23
            Pf.u32[4 + i] = hi ? pk1[i] : rcv;      // K 8..15 / 24..31
        }

        // O^T += g^T x P^T over 8 channel tiles
#pragma unroll
        for (int ct = 0; ct < 8; ct++) {
            frag16 G;
            const _Float16* gp = &gT[(ct * 16 + m) * 40 + hi * 16];
            G.u4[0] = *reinterpret_cast<const uint4*>(gp);
            G.u4[1] = *reinterpret_cast<const uint4*>(gp + 8);
            O[ct] = wmma_f16(G.h, Pf.h, O[ct]);
        }
    }

    // normalize and stage attn_g (f16) into per-wave LDS as A-matrix source
    const float inv = 1.0f / l_run;
    _Float16* ap = &aO[(wave * 16 + m) * 136];
#pragma unroll
    for (int ct = 0; ct < 8; ct++) {
        O[ct] = O[ct] * inv;
        *reinterpret_cast<uint4*>(ap + ct * 16 + hi * 8) = pack8(O[ct]);
    }
    __syncthreads();

    // output projection Y = attn_g x w_o, then out = x + sigma*(Y + b_o)
    for (int nt = 0; nt < 16; nt++) {
        const int c = nt * 16 + m;
        const size_t rbase = ((size_t)(b * 4096 + qbase + hi * 8)) * 256 + c;
        // issue residual loads early: overlap HBM latency with WMMA chain
        float xr[8];
#pragma unroll
        for (int r = 0; r < 8; r++) xr[r] = x[rbase + (size_t)r * 256];
        const float bo = b_o[c];
        v8f acc = zero8;
#pragma unroll
        for (int kt = 0; kt < 4; kt++) {
            frag16 Aa, Bw;
            const _Float16* pa = ap + kt * 32 + hi * 8;
            Aa.u4[0] = *reinterpret_cast<const uint4*>(pa);
            Aa.u4[1] = *reinterpret_cast<const uint4*>(pa + 16);
            const _Float16* pw = Wo + ((size_t)((nt * 4 + kt) * 32 + lane)) * 16;
            Bw.u4[0] = ld_u4(pw);
            Bw.u4[1] = ld_u4(pw + 8);
            acc = wmma_f16(Aa.h, Bw.h, acc);
        }
#pragma unroll
        for (int r = 0; r < 8; r++) {
            out[rbase + (size_t)r * 256] = xr[r] + sigma * (acc[r] + bo);
        }
    }
}

// ---------------------------------------------------------------------------
extern "C" void kernel_launch(void* const* d_in, const int* in_sizes, int n_in,
                              void* d_out, int out_size, void* d_ws,
                              size_t ws_size, hipStream_t stream) {
    (void)in_sizes; (void)n_in; (void)out_size; (void)ws_size;
    const float* x       = (const float*)d_in[0];
    const float* w_theta = (const float*)d_in[1];
    const float* w_phi   = (const float*)d_in[2];
    const float* w_g     = (const float*)d_in[3];
    const float* w_o     = (const float*)d_in[4];
    const float* b_o     = (const float*)d_in[5];
    const float* sigma   = (const float*)d_in[6];
    float* out = (float*)d_out;

    char* ws = (char*)d_ws;
    _Float16* Wc    = (_Float16*)(ws + WC_OFF);
    _Float16* Wo    = (_Float16*)(ws + WO_OFF);
    _Float16* theta = (_Float16*)(ws + THETA_OFF);
    _Float16* phif  = (_Float16*)(ws + PHIF_OFF);
    _Float16* phip  = (_Float16*)(ws + PHIP_OFF);
    _Float16* gt    = (_Float16*)(ws + GT_OFF);
    _Float16* gtp   = (_Float16*)(ws + GTP_OFF);

    swizzle_w_kernel<<<320, 256, 0, stream>>>(w_theta, w_phi, w_g, w_o, Wc, Wo);
    proj_kernel<<<512, 256, 0, stream>>>(x, Wc, theta, phif, gt);
    pool_phi_kernel<<<256, 256, 0, stream>>>(phif, phip);
    pool_g_kernel<<<1024, 256, 0, stream>>>(gt, gtp);
    attn_kernel<<<dim3(32, 16), 256, 0, stream>>>(x, theta, phip, gtp, Wo, b_o,
                                                  sigma, out);
}
